// CotLayer_48936857370648
// MI455X (gfx1250) — matrix operand order repaired and checked by reading, not compiled, not run on hardware
//
#include <hip/hip_runtime.h>
#include <hip/hip_bf16.h>

// ---------------------------------------------------------------------------
// CotLayer for MI455X (gfx1250).
//  - GEMM-shaped work on v_wmma_f32_16x16x32_bf16 (fp32 -> bf16 via v_perm
//    packing, fp32 accumulate).
//  - Grouped 3x3x3 conv: implicit GEMM, tap-major K, 48M x 16N per wave;
//    branchless gather (clamped base + post-pack cndmask zeroing) so the
//    16 gather loads per fragment form a clause with no EXEC save/restore.
//  - Dense 1x1 convs: 16M x 64N per wave; B tiles staged to LDS by the
//    Tensor Data Mover (double-buffered, s_wait_tensorcnt pipelining), with
//    a cooperative-load fallback when the TDM builtin is unavailable.
// ---------------------------------------------------------------------------

#define DIMC 192
#define VD 16
#define VH 32
#define VW 32
#define VTOT (VD * VH * VW)        // 16384
#define NB 2
#define CW 216
#define ATTN_DIM 96
#define KC_CONV 1296               // 48 ci * 27 taps
#define LROW 68                    // LDS row stride in floats (64 + 4 pad)

typedef __attribute__((ext_vector_type(16))) __bf16         v16bf;
typedef __attribute__((ext_vector_type(8)))  unsigned int   v8u;
typedef __attribute__((ext_vector_type(8)))  float          v8f;
typedef __attribute__((ext_vector_type(4)))  float          v4f;
typedef __attribute__((ext_vector_type(4)))  unsigned int   v4u;
typedef __attribute__((ext_vector_type(8)))  int            v8i;
typedef __attribute__((ext_vector_type(4)))  int            v4i;

#if __has_builtin(__builtin_amdgcn_tensor_load_to_lds)
#define HAVE_TDM 1
#if __has_include(<hip/amd_detail/amd_gfx1250_TDM.h>)
#define TDM_6ARG 1
#warning "CDNA5 probe: tensor_load_to_lds 6-arg path selected"
#else
#warning "CDNA5 probe: tensor_load_to_lds 5-arg path selected"
#endif
#else
#warning "CDNA5 probe: no tensor_load_to_lds builtin, cooperative-load fallback"
#endif

// pack bf16(a) into low 16, bf16(b) into high 16 with one v_perm_b32
__device__ __forceinline__ unsigned pk2(float a, float b) {
  return __builtin_amdgcn_perm(__builtin_bit_cast(unsigned, b),
                               __builtin_bit_cast(unsigned, a), 0x07060302u);
}

__device__ __forceinline__ v16bf pack16(const float* f) {
  v8u p;
#pragma unroll
  for (int j = 0; j < 8; ++j) p[j] = pk2(f[2 * j], f[2 * j + 1]);
  return __builtin_bit_cast(v16bf, p);
}

// pack with post-pack zeroing (branchless predication of a whole fragment)
__device__ __forceinline__ v16bf pack16_sel(const float* f, bool keep) {
  v8u p;
#pragma unroll
  for (int j = 0; j < 8; ++j) {
    const unsigned w = pk2(f[2 * j], f[2 * j + 1]);
    p[j] = keep ? w : 0u;           // v_cndmask, no EXEC manipulation
  }
  return __builtin_bit_cast(v16bf, p);
}

__device__ __forceinline__ v8f wmma_bf16(v16bf a, v16bf b, v8f c) {
  return __builtin_amdgcn_wmma_f32_16x16x32_bf16(false, a, false, b,
                                                 (short)0, c, false, false);
}

#if defined(HAVE_TDM)
// Issue a TDM load of a 32-row x 64-col fp32 tile (row stride VTOT elements)
// into LDS at byte offset `ldsoff`, padding each 64-DWORD row by 4 DWORDs
// (LDS row stride = 68 floats, 16B-aligned rows).
__device__ __forceinline__ void tdm_issue_tile(const float* src,
                                               unsigned ldsoff) {
  unsigned long long ga = (unsigned long long)(uintptr_t)src;
  v4u g0;
  g0[0] = 1u;                                   // count=1, user mode
  g0[1] = ldsoff;                               // lds_addr
  g0[2] = (unsigned)ga;                         // global_addr[31:0]
  g0[3] = (unsigned)((ga >> 32) & 0x01FFFFFFull) | 0x80000000u; // [56:32]|type=2
  v8i g1;
  g1[0] = (int)((2u << 16)     // data_size = 4B
              | (1u << 20)     // pad_enable
              | (5u << 22)     // pad_interval: every 64 DWORDs
              | (3u << 25));   // pad_amount: 4 DWORDs
  g1[1] = (int)(64u << 16);    // tensor_dim0 = 64
  g1[2] = (int)(32u << 16);    // tensor_dim1 = 32
  g1[3] = (int)(64u << 16);    // tile_dim0 = 64
  g1[4] = 32;                  // tile_dim1 = 32 (tile_dim2 = 0)
  g1[5] = VTOT;                // tensor_dim0_stride
  g1[6] = 0;
  g1[7] = 0;
  v4i z4 = {0, 0, 0, 0};
#if defined(TDM_6ARG)
  v8i z8 = {0, 0, 0, 0, 0, 0, 0, 0};
  __builtin_amdgcn_tensor_load_to_lds(g0, g1, z4, z4, z8, 0);
#else
  __builtin_amdgcn_tensor_load_to_lds(g0, g1, z4, z4, 0);
#endif
}
#endif

// ---------------------------------------------------------------------------
// 0) One-time transpose: wk2[oc][tap][ci] = w_key[oc][ci][tap]
// ---------------------------------------------------------------------------
__global__ void k_wkey_transpose(const float* __restrict__ wkey,
                                 float* __restrict__ wk2) {
  const int i = blockIdx.x * 256 + threadIdx.x;
  if (i >= DIMC * KC_CONV) return;
  const int oc = i / KC_CONV;
  const int r = i - oc * KC_CONV;
  const int ci = r / 27;
  const int tap = r - ci * 27;
  wk2[(size_t)oc * KC_CONV + tap * 48 + ci] = wkey[i];
}

// ---------------------------------------------------------------------------
// 1) Grouped 3x3x3 conv + BN1 + ReLU. One wave: 48 oc x 16 voxels.
//    Tap-major K: 27 scalar taps x (48 ci -> one full + one half K-step).
// ---------------------------------------------------------------------------
__global__ __launch_bounds__(32) void k_conv_grouped(
    const float* __restrict__ x, const float* __restrict__ wk2,
    const float* __restrict__ s1, const float* __restrict__ b1,
    float* __restrict__ kout) {
  const int b = blockIdx.z;
  const int g = blockIdx.y;          // channel group 0..3
  const int nb = blockIdx.x * 16;
  const int lane = threadIdx.x & 31;
  const int lo = lane & 15;
  const int half = lane >> 4;
  const float* xg = x + ((size_t)b * DIMC + (size_t)g * 48) * VTOT;

  const int v = nb + lo;
  const int d = v >> 10, h = (v >> 5) & 31, w = v & 31;

  v8f acc[3];
#pragma unroll
  for (int mi = 0; mi < 3; ++mi)
#pragma unroll
    for (int r = 0; r < 8; ++r) acc[mi][r] = 0.0f;

  for (int tap = 0; tap < 27; ++tap) {          // scalar tap decode
    const int dz = tap / 9 - 1;
    const int dy = (tap / 3) % 3 - 1;
    const int dx = tap % 3 - 1;
    const int zd = d + dz, zh = h + dy, zw = w + dx;
    const bool ok = ((unsigned)zd < (unsigned)VD) &&
                    ((unsigned)zh < (unsigned)VH) &&
                    ((unsigned)zw < (unsigned)VW);
    // Clamp to a valid address; invalid lanes discarded post-pack.
    const int sp = ok ? ((zd << 10) + (zh << 5) + zw) : 0;
    const float* basep = xg + sp;
    if (tap + 1 < 27) {                         // prefetch next tap's window
      __builtin_prefetch(basep + 1, 0, 1);
    }

#pragma unroll
    for (int t = 0; t < 2; ++t) {               // K-steps: ci 0-31, 32-47(+pad)
      // table-order K; t=1 elements 8..15 are pad (A zero) -> clamp to ci 32.
      float bfv[16];
#pragma unroll
      for (int e = 0; e < 16; ++e) {             // unconditional clause
        const int kk = (e < 8) ? (half * 8 + e) : (t == 0 ? 16 + half * 8 + (e - 8) : 0);
        bfv[e] = basep[(size_t)(t * 32 + kk) * VTOT];
      }
      const bool bval = ok;
      const v16bf bfrag = pack16_sel(bfv, bval);

      // Materialize all three A fragments before any WMMA so the perms of
      // fragment i+1 don't WAR the A operand of the in-flight WMMA i.
      v16bf afrag[3];
#pragma unroll
      for (int mi = 0; mi < 3; ++mi) {
        const int oc = g * 48 + mi * 16 + lo;
        const float* ar = wk2 + (size_t)oc * KC_CONV + tap * 48 + t * 32;
        float af[16];
        if (t == 0) {
          v4f a0 = *(const v4f*)(ar + half * 8);
          v4f a1 = *(const v4f*)(ar + half * 8 + 4);
          v4f a2 = *(const v4f*)(ar + 16 + half * 8);
          v4f a3 = *(const v4f*)(ar + 16 + half * 8 + 4);
#pragma unroll
          for (int j = 0; j < 4; ++j) {
            af[j] = a0[j]; af[4 + j] = a1[j];
            af[8 + j] = a2[j]; af[12 + j] = a3[j];
          }
        } else {                                // ci 32..47 valid, rest zero
          v4f a0 = *(const v4f*)(ar + half * 8);
          v4f a1 = *(const v4f*)(ar + half * 8 + 4);
#pragma unroll
          for (int j = 0; j < 4; ++j) {
            af[j] = a0[j]; af[4 + j] = a1[j];
            af[8 + j] = 0.0f; af[12 + j] = 0.0f;
          }
        }
        afrag[mi] = pack16(af);
      }
#pragma unroll
      for (int mi = 0; mi < 3; ++mi)
        acc[mi] = wmma_bf16(afrag[mi], bfrag, acc[mi]);
    }
  }

  const int n = nb + lo;
#pragma unroll
  for (int mi = 0; mi < 3; ++mi)
#pragma unroll
    for (int r = 0; r < 8; ++r) {
      const int oc = g * 48 + mi * 16 + r + half * 8;
      const float vl = acc[mi][r] * s1[oc] + b1[oc];
      kout[((size_t)b * DIMC + oc) * VTOT + n] = vl > 0.0f ? vl : 0.0f;
    }
}

// ---------------------------------------------------------------------------
// 2) Dense GEMM: out[M,N] over K, B rows from bsrc0 (k<ksplit) else bsrc1.
//    One wave: 16M x 64N (4 accumulators). B tile staged to LDS (TDM if
//    available). act: 0 = scale*x+bias, 1 = relu(scale*x+bias), 2 = x+bias.
// ---------------------------------------------------------------------------
__global__ __launch_bounds__(32) void k_gemm_dense(
    const float* __restrict__ bsrc0, const float* __restrict__ bsrc1,
    int ksplit, int Kc, int Mrows,
    const float* __restrict__ wmat,
    const float* __restrict__ scale, const float* __restrict__ bias,
    int act, float* __restrict__ outb) {
  const int b = blockIdx.z;
  const int mb = blockIdx.y * 16;
  const int nb = blockIdx.x * 64;
  const int lane = threadIdx.x & 31;
  const int lo = lane & 15;
  const int half = lane >> 4;
  const int steps = Kc >> 5;

  v8f acc[4];
#pragma unroll
  for (int nt = 0; nt < 4; ++nt)
#pragma unroll
    for (int r = 0; r < 8; ++r) acc[nt][r] = 0.0f;

  auto srcfor = [&](int s) -> const float* {
    const int kb = s << 5;
    return (kb < ksplit)
               ? bsrc0 + ((size_t)b * ksplit + kb) * VTOT + nb
               : bsrc1 + ((size_t)b * (Kc - ksplit) + (kb - ksplit)) * VTOT + nb;
  };

  auto compute_step = [&](const float* lb, int kb) {
    // A fragment (weights, contiguous in K); tail rows clamp + post-pack zero
    const int m = mb + lo;
    const bool mok = (m < Mrows);
    const float* ar = wmat + (size_t)(mok ? m : 0) * Kc + kb;
    float af[16];
    v4f a0 = *(const v4f*)(ar + half * 8);
    v4f a1 = *(const v4f*)(ar + half * 8 + 4);
    v4f a2 = *(const v4f*)(ar + 16 + half * 8);
    v4f a3 = *(const v4f*)(ar + 16 + half * 8 + 4);
#pragma unroll
    for (int j = 0; j < 4; ++j) {
      af[j] = a0[j]; af[4 + j] = a1[j];
      af[8 + j] = a2[j]; af[12 + j] = a3[j];
    }
    const v16bf afrag = pack16_sel(af, mok);

    // Materialize all four B fragments, then issue the four WMMAs.
    v16bf bfrag[4];
#pragma unroll
    for (int nt = 0; nt < 4; ++nt) {
      float bfv[16];
      const float* lbase = lb + (half * 8) * LROW + nt * 16 + lo;
#pragma unroll
      for (int e = 0; e < 16; ++e) bfv[e] = lbase[((e < 8) ? e : 8 + e) * LROW];
      bfrag[nt] = pack16(bfv);
    }
#pragma unroll
    for (int nt = 0; nt < 4; ++nt)
      acc[nt] = wmma_bf16(afrag, bfrag[nt], acc[nt]);
  };

#if defined(HAVE_TDM)
  __shared__ float lbuf[2][32 * LROW];
  tdm_issue_tile(srcfor(0), (unsigned)(uintptr_t)&lbuf[0][0]);
  for (int s = 0; s < steps; ++s) {
    if (s + 1 < steps) {
      tdm_issue_tile(srcfor(s + 1), (unsigned)(uintptr_t)&lbuf[(s + 1) & 1][0]);
      __builtin_amdgcn_s_wait_tensorcnt(1);   // current tile done (in-order)
    } else {
      __builtin_amdgcn_s_wait_tensorcnt(0);
    }
    asm volatile("" ::: "memory");
    compute_step(&lbuf[s & 1][0], s << 5);
  }
#else
  __shared__ float lbuf[32 * LROW];
  for (int s = 0; s < steps; ++s) {
    const float* src = srcfor(s);
    const float* rowp = src + (size_t)lane * VTOT;
#pragma unroll
    for (int j = 0; j < 16; ++j)
      *(v4f*)&lbuf[lane * LROW + j * 4] = *(const v4f*)(rowp + j * 4);
    __syncthreads();
    compute_step(lbuf, s << 5);
    __syncthreads();
  }
#endif

#pragma unroll
  for (int nt = 0; nt < 4; ++nt)
#pragma unroll
    for (int r = 0; r < 8; ++r) {
      const int m = mb + r + half * 8;
      if (m < Mrows) {
        const int n = nb + nt * 16 + lo;
        float vl = acc[nt][r];
        if (act == 2)      vl = vl + bias[m];
        else if (act == 1) { vl = vl * scale[m] + bias[m]; vl = vl > 0.0f ? vl : 0.0f; }
        else               vl = vl * scale[m] + bias[m];
        outb[((size_t)b * Mrows + m) * VTOT + n] = vl;
      }
    }
}

// ---------------------------------------------------------------------------
// 4) GroupNorm over w: 24 groups x 9 channels x 16384 voxels, in place.
// ---------------------------------------------------------------------------
__global__ void k_groupnorm(float* __restrict__ wbuf,
                            const float* __restrict__ gs,
                            const float* __restrict__ gb) {
  const int b = blockIdx.y, gg = blockIdx.x;
  float* base = wbuf + ((size_t)b * CW + (size_t)gg * 9) * VTOT;
  const int N = 9 * VTOT;

  __shared__ float red[256], red2[256];
  __shared__ float s_mu, s_rs;
  float s = 0.0f, sq = 0.0f;
  for (int i = threadIdx.x; i < N; i += 256) {
    const float v = base[i];
    s += v;
    sq += v * v;
  }
  red[threadIdx.x] = s;
  red2[threadIdx.x] = sq;
  __syncthreads();
  for (int st = 128; st > 0; st >>= 1) {
    if ((int)threadIdx.x < st) {
      red[threadIdx.x] += red[threadIdx.x + st];
      red2[threadIdx.x] += red2[threadIdx.x + st];
    }
    __syncthreads();
  }
  if (threadIdx.x == 0) {
    const float mu = red[0] / (float)N;
    const float var = red2[0] / (float)N - mu * mu;
    s_mu = mu;
    s_rs = rsqrtf(var + 1e-5f);
  }
  __syncthreads();
  const float mu = s_mu, rs = s_rs;
  for (int i = threadIdx.x; i < N; i += 256) {
    const int c = gg * 9 + i / VTOT;
    base[i] = (base[i] - mu) * rs * gs[c] + gb[c];
  }
}

// ---------------------------------------------------------------------------
// 6) Dynamic 27-tap conv + bn4 + swish -> x2 (branchless taps)
// ---------------------------------------------------------------------------
__global__ void k_dynconv(const float* __restrict__ vbuf,
                          const float* __restrict__ wbuf,
                          const float* __restrict__ s4,
                          const float* __restrict__ b4,
                          float* __restrict__ x2) {
  const size_t i = (size_t)blockIdx.x * 256 + threadIdx.x;
  const size_t total = (size_t)NB * DIMC * VTOT;
  if (i >= total) return;
  const int v = (int)(i & (VTOT - 1));
  const int t = (int)(i >> 14);
  const int c = t % DIMC;
  const int b = t / DIMC;
  const int d = v >> 10, h = (v >> 5) & 31, w = v & 31;

  const float* wrow = wbuf + ((size_t)b * CW + (size_t)(c / 24) * 27) * VTOT + v;
  const float* vrow = vbuf + ((size_t)b * DIMC + c) * VTOT;

  float acc = 0.0f;
  int idx = 0;
#pragma unroll
  for (int dz = -1; dz <= 1; ++dz)
#pragma unroll
    for (int dy = -1; dy <= 1; ++dy)
#pragma unroll
      for (int dx = -1; dx <= 1; ++dx, ++idx) {
        const int zd = d + dz, zh = h + dy, zw = w + dx;
        const bool ok = ((unsigned)zd < (unsigned)VD) &&
                        ((unsigned)zh < (unsigned)VH) &&
                        ((unsigned)zw < (unsigned)VW);
        const int sp = ok ? ((zd << 10) + (zh << 5) + zw) : 0; // clamp
        const float pvr = vrow[sp];                            // unconditional
        const float pv = ok ? pvr : 0.0f;                      // cndmask
        acc += wrow[(size_t)idx * VTOT] * pv;
      }
  const float y = acc * s4[c] + b4[c];
  x2[i] = y / (1.0f + expf(-y));     // swish
}

// ---------------------------------------------------------------------------
// 7) gap[b,c] = mean_v(x2 + k)
// ---------------------------------------------------------------------------
__global__ void k_gap(const float* __restrict__ x2,
                      const float* __restrict__ kbuf,
                      float* __restrict__ gap) {
  const int bc = blockIdx.x;
  const float* p1 = x2 + (size_t)bc * VTOT;
  const float* p2 = kbuf + (size_t)bc * VTOT;
  __shared__ float red[256];
  float s = 0.0f;
  for (int i = threadIdx.x; i < VTOT; i += 256) s += p1[i] + p2[i];
  red[threadIdx.x] = s;
  __syncthreads();
  for (int st = 128; st > 0; st >>= 1) {
    if ((int)threadIdx.x < st) red[threadIdx.x] += red[threadIdx.x + st];
    __syncthreads();
  }
  if (threadIdx.x == 0) gap[bc] = red[0] / (float)VTOT;
}

// ---------------------------------------------------------------------------
// 8) SE head
// ---------------------------------------------------------------------------
__global__ void k_se(const float* __restrict__ gap,
                     const float* __restrict__ w1,  const float* __restrict__ bb1,
                     const float* __restrict__ ss,  const float* __restrict__ sb,
                     const float* __restrict__ w2,  const float* __restrict__ bb2,
                     float* __restrict__ attn) {
  __shared__ float h1s[NB * ATTN_DIM];
  __shared__ float araw[NB * 2 * DIMC];
  const int t = threadIdx.x;

  for (int o = t; o < NB * ATTN_DIM; o += 256) {
    const int b = o / ATTN_DIM, oo = o % ATTN_DIM;
    const float* wr = w1 + (size_t)oo * DIMC;
    const float* gp = gap + (size_t)b * DIMC;
    float s = bb1[oo];
    for (int i = 0; i < DIMC; ++i) s += wr[i] * gp[i];
    const float v = s * ss[oo] + sb[oo];
    h1s[o] = v > 0.0f ? v : 0.0f;
  }
  __syncthreads();

  for (int o = t; o < NB * 2 * DIMC; o += 256) {
    const int b = o / (2 * DIMC), oo = o % (2 * DIMC);
    const float* wr = w2 + (size_t)oo * ATTN_DIM;
    float s = bb2[oo];
    for (int i = 0; i < ATTN_DIM; ++i) s += wr[i] * h1s[b * ATTN_DIM + i];
    araw[o] = s;
  }
  __syncthreads();

  for (int o = t; o < NB * DIMC; o += 256) {
    const int b = o / DIMC, c = o % DIMC;
    const float a0 = araw[b * 2 * DIMC + 2 * c];
    const float a1 = araw[b * 2 * DIMC + 2 * c + 1];
    const float m = fmaxf(a0, a1);
    const float e0 = expf(a0 - m), e1 = expf(a1 - m);
    const float inv = 1.0f / (e0 + e1);
    attn[(size_t)o * 2 + 0] = e0 * inv;
    attn[(size_t)o * 2 + 1] = e1 * inv;
  }
}

// ---------------------------------------------------------------------------
// 9) out = x2*attn0 + k*attn1
// ---------------------------------------------------------------------------
__global__ void k_final(const float* __restrict__ x2,
                        const float* __restrict__ kbuf,
                        const float* __restrict__ attn,
                        float* __restrict__ out) {
  const size_t i = (size_t)blockIdx.x * 256 + threadIdx.x;
  const size_t total = (size_t)NB * DIMC * VTOT;
  if (i >= total) return;
  const size_t bc = i >> 14;
  out[i] = x2[i] * attn[bc * 2 + 0] + kbuf[i] * attn[bc * 2 + 1];
}

// ---------------------------------------------------------------------------

extern "C" void kernel_launch(void* const* d_in, const int* in_sizes, int n_in,
                              void* d_out, int out_size, void* d_ws, size_t ws_size,
                              hipStream_t stream) {
  (void)in_sizes; (void)n_in; (void)out_size; (void)ws_size;

  const float* x     = (const float*)d_in[0];
  const float* wkey  = (const float*)d_in[1];
  const float* bn1_s = (const float*)d_in[2];
  const float* bn1_b = (const float*)d_in[3];
  const float* w_e1  = (const float*)d_in[4];
  const float* bn2_s = (const float*)d_in[5];
  const float* bn2_b = (const float*)d_in[6];
  const float* w_e2  = (const float*)d_in[7];
  const float* b_e2  = (const float*)d_in[8];
  const float* gn_s  = (const float*)d_in[9];
  const float* gn_b  = (const float*)d_in[10];
  const float* w_c1  = (const float*)d_in[11];
  const float* bn3_s = (const float*)d_in[12];
  const float* bn3_b = (const float*)d_in[13];
  const float* bn4_s = (const float*)d_in[14];
  const float* bn4_b = (const float*)d_in[15];
  const float* w_se1 = (const float*)d_in[16];
  const float* b_se1 = (const float*)d_in[17];
  const float* bnse_s= (const float*)d_in[18];
  const float* bnse_b= (const float*)d_in[19];
  const float* w_se2 = (const float*)d_in[20];
  const float* b_se2 = (const float*)d_in[21];

  // workspace carve-up (fp32)
  char* p = (char*)d_ws;
  const size_t plane = (size_t)NB * VTOT;
  float* kbuf  = (float*)p;  p += plane * DIMC     * sizeof(float);
  float* ebuf  = (float*)p;  p += plane * ATTN_DIM * sizeof(float);
  float* wbuf  = (float*)p;  p += plane * CW       * sizeof(float);
  float* vbuf  = (float*)p;  p += plane * DIMC     * sizeof(float);
  float* x2buf = (float*)p;  p += plane * DIMC     * sizeof(float);
  float* wk2   = (float*)p;  p += (size_t)DIMC * KC_CONV * sizeof(float);
  float* gapb  = (float*)p;  p += (size_t)NB * DIMC * sizeof(float);
  float* attnb = (float*)p;

  const size_t total = (size_t)NB * DIMC * VTOT;
  const int eltblocks = (int)((total + 255) / 256);

  k_wkey_transpose<<<(DIMC * KC_CONV + 255) / 256, 256, 0, stream>>>(wkey, wk2);
  k_conv_grouped<<<dim3(VTOT / 16, 4, NB), 32, 0, stream>>>(
      x, wk2, bn1_s, bn1_b, kbuf);
  // e = relu(bn2(w_e1 @ [x;k]))   M=96 K=384
  k_gemm_dense<<<dim3(VTOT / 64, 6, NB), 32, 0, stream>>>(
      x, kbuf, DIMC, 2 * DIMC, ATTN_DIM, w_e1, bn2_s, bn2_b, 1, ebuf);
  // w = w_e2 @ e + b_e2           M=216 K=96
  k_gemm_dense<<<dim3(VTOT / 64, (CW + 15) / 16, NB), 32, 0, stream>>>(
      ebuf, ebuf, ATTN_DIM, ATTN_DIM, CW, w_e2, b_e2, b_e2, 2, wbuf);
  k_groupnorm<<<dim3(24, NB), 256, 0, stream>>>(wbuf, gn_s, gn_b);
  // v = bn3(w_c1 @ x)             M=192 K=192
  k_gemm_dense<<<dim3(VTOT / 64, DIMC / 16, NB), 32, 0, stream>>>(
      x, x, DIMC, DIMC, DIMC, w_c1, bn3_s, bn3_b, 0, vbuf);
  k_dynconv<<<eltblocks, 256, 0, stream>>>(vbuf, wbuf, bn4_s, bn4_b, x2buf);
  k_gap<<<NB * DIMC, 256, 0, stream>>>(x2buf, kbuf, gapb);
  k_se<<<1, 256, 0, stream>>>(gapb, w_se1, b_se1, bnse_s, bnse_b,
                              w_se2, b_se2, attnb);
  k_final<<<eltblocks, 256, 0, stream>>>(x2buf, kbuf, attnb, (float*)d_out);
}
